// MultiHeadAttention_44203803411006
// MI455X (gfx1250) — matrix operand order repaired and checked by reading, not compiled, not run on hardware
//
#include <hip/hip_runtime.h>

typedef __attribute__((ext_vector_type(16))) __bf16 v16bf;
typedef __attribute__((ext_vector_type(8)))  float  v8f;

union FragBF16 {
  v16bf v;
  unsigned int u[8];
};

#define B_  4
#define S_  2048
#define D_  1024
#define H_  16
#define DK_ 64
#define DV_ 64
#define M_  (B_ * S_)      // 8192 rows
#define N_  (H_ * DK_)     // 1024 cols
#define K_  D_             // 1024 reduction

// ---------------- GEMM tiling ----------------
#define BM 128
#define BN 64
#define BK 32
#define BKP (BK + 4)       // padded LDS row (36 elems = 72B, dword aligned)

// 16-byte async copy global->LDS (ASYNCcnt-tracked).  LDS dst must be 16B aligned.
__device__ __forceinline__ void async_copy16(const void* gptr, void* lptr) {
  const unsigned lds_off = (unsigned)(size_t)lptr;  // shared-aperture low 32 bits = LDS offset
  const unsigned long long gaddr = (unsigned long long)(size_t)gptr;
  asm volatile("global_load_async_to_lds_b128 %0, %1, off"
               :: "v"(lds_off), "v"(gaddr)
               : "memory");
}
__device__ __forceinline__ void wait_async0() {
  asm volatile("s_wait_asynccnt 0x0" ::: "memory");
}
__device__ __forceinline__ void wait_ds0() {
  asm volatile("s_wait_dscnt 0x0" ::: "memory");
}

// C = A[M,K] * W[K,N] + bias.  SPLIT: write bf16 in [B,H,S,64]; else f32 [M,N].
// Double-buffered LDS: stage tile k+1 while computing tile k.
template <bool SPLIT_HEADS_BF16>
__global__ __launch_bounds__(256)
void gemm_wmma_kernel(const float* __restrict__ A,
                      const float* __restrict__ W,
                      const float* __restrict__ bias,
                      float* __restrict__ outF,
                      unsigned short* __restrict__ outB)
{
  __shared__ __attribute__((aligned(16))) __bf16 As[2][BM][BKP];
  __shared__ __attribute__((aligned(16))) __bf16 Ws[2][BN][BKP];  // transposed [n][k]

  const int tid  = threadIdx.x;
  const int lane = tid & 31;
  const int half = lane >> 4;
  const int ln   = lane & 15;
  const int wid  = tid >> 5;
  const int wm   = wid >> 1;   // 0..3 : 32-row slab
  const int wn   = wid & 1;    // 0..1 : 32-col slab
  const int m0   = blockIdx.y * BM;
  const int n0   = blockIdx.x * BN;

  v8f acc[2][2] = {};

  const int ar  = tid >> 3;          // A stage: row 0..31 (x4 passes)
  const int ac  = (tid & 7) << 2;    // A stage: col (float4)
  const int wkr = tid & 31;          // W stage: k row
  const int wnc = (tid >> 5) << 3;   // W stage: 8 n cols

  auto stage = [&](int k0, int buf) {
#pragma unroll
    for (int p = 0; p < 4; ++p) {
      const int row = ar + 32 * p;
      const float4 f = *(const float4*)&A[(size_t)(m0 + row) * K_ + k0 + ac];
      As[buf][row][ac + 0] = (__bf16)f.x;
      As[buf][row][ac + 1] = (__bf16)f.y;
      As[buf][row][ac + 2] = (__bf16)f.z;
      As[buf][row][ac + 3] = (__bf16)f.w;
    }
    const float4 f0 = *(const float4*)&W[(size_t)(k0 + wkr) * N_ + n0 + wnc];
    const float4 f1 = *(const float4*)&W[(size_t)(k0 + wkr) * N_ + n0 + wnc + 4];
    Ws[buf][wnc + 0][wkr] = (__bf16)f0.x;
    Ws[buf][wnc + 1][wkr] = (__bf16)f0.y;
    Ws[buf][wnc + 2][wkr] = (__bf16)f0.z;
    Ws[buf][wnc + 3][wkr] = (__bf16)f0.w;
    Ws[buf][wnc + 4][wkr] = (__bf16)f1.x;
    Ws[buf][wnc + 5][wkr] = (__bf16)f1.y;
    Ws[buf][wnc + 6][wkr] = (__bf16)f1.z;
    Ws[buf][wnc + 7][wkr] = (__bf16)f1.w;
  };

  constexpr int NIT = K_ / BK;
  stage(0, 0);

  for (int it = 0; it < NIT; ++it) {
    __syncthreads();
    if (it + 1 < NIT) stage((it + 1) * BK, (it + 1) & 1);
    const int cur = it & 1;

    // A fragment: lane = M row; VGPR j -> K pair 2(j&3)+16(j>>2), +8 for hi half-lanes
    FragBF16 af[2], bfr[2];
#pragma unroll
    for (int i = 0; i < 2; ++i) {
      const int m = wm * 32 + i * 16 + ln;
#pragma unroll
      for (int j = 0; j < 8; ++j) {
        const int kb = 2 * (j & 3) + ((j >> 2) << 4) + (half << 3);
        af[i].u[j] = *(const unsigned int*)&As[cur][m][kb];
      }
    }
    // B fragment: lane = N col; VGPR j -> K pair 2(j&3)+16(j>>2), +8 for hi half-lanes
#pragma unroll
    for (int jt = 0; jt < 2; ++jt) {
      const int n = wn * 32 + jt * 16 + ln;
#pragma unroll
      for (int j = 0; j < 8; ++j) {
        const int kb = 2 * (j & 3) + ((j >> 2) << 4) + (half << 3);
        bfr[jt].u[j] = *(const unsigned int*)&Ws[cur][n][kb];
      }
    }
#pragma unroll
    for (int i = 0; i < 2; ++i)
#pragma unroll
      for (int jt = 0; jt < 2; ++jt)
        acc[i][jt] = __builtin_amdgcn_wmma_f32_16x16x32_bf16(
            false, af[i].v, false, bfr[jt].v, (short)0, acc[i][jt], false, false);
  }

  // Epilogue.  C layout: VGPR r -> row r + 8*half, col = lane&15.
#pragma unroll
  for (int i = 0; i < 2; ++i) {
#pragma unroll
    for (int jt = 0; jt < 2; ++jt) {
      const int gn = n0 + wn * 32 + jt * 16 + ln;
      const float bv = bias[gn];
#pragma unroll
      for (int r = 0; r < 8; ++r) {
        const int gm = m0 + wm * 32 + i * 16 + r + 8 * half;
        const float val = acc[i][jt][r] + bv;
        if constexpr (SPLIT_HEADS_BF16) {
          const int bb = gm >> 11;            // / S_
          const int s  = gm & (S_ - 1);
          const int hh = gn >> 6;             // / DK_
          const int d  = gn & (DK_ - 1);
          const size_t idx = (((size_t)bb * H_ + hh) * S_ + s) * DK_ + d;
          __bf16 t = (__bf16)val;
          outB[idx] = __builtin_bit_cast(unsigned short, t);
        } else {
          outF[(size_t)gm * N_ + gn] = val;
        }
      }
    }
  }
}

// ------- Flash attention: 256 threads = 8 waves, 128 query rows per block ----
// K/V chunks staged cooperatively (K via async-to-LDS), double-buffered.
__global__ __launch_bounds__(256)
void attn_wmma_kernel(const unsigned short* __restrict__ Qu,
                      const unsigned short* __restrict__ Ku,
                      const unsigned short* __restrict__ Vu,
                      const float* __restrict__ mask,
                      float* __restrict__ ctx)
{
  constexpr int QP = DK_ + 8;   // 72 elems = 144B rows (16B aligned)
  constexpr int NC = S_ / 32;   // 64 key chunks
  __shared__ __attribute__((aligned(16))) __bf16 Qs[128][QP];
  __shared__ __attribute__((aligned(16))) __bf16 Ks[2][32][QP];
  __shared__ __attribute__((aligned(16))) __bf16 Vt[2][DV_][BKP];  // [d][key]
  __shared__ __attribute__((aligned(16))) __bf16 Pt[8][16][BKP];   // per-wave

  const int tid  = threadIdx.x;
  const int lane = tid & 31;
  const int wid  = tid >> 5;
  const int half = lane >> 4;
  const int ln   = lane & 15;
  const int qb0  = blockIdx.x * 128;
  const int h    = blockIdx.y;
  const int b    = blockIdx.z;

  const __bf16* Q  = (const __bf16*)Qu + (((size_t)b * H_ + h) * S_ + qb0) * DK_;
  const __bf16* Kp = (const __bf16*)Ku + (((size_t)b * H_ + h) * S_) * DK_;
  const __bf16* Vp = (const __bf16*)Vu + (((size_t)b * H_ + h) * S_) * DK_;

  // K chunk async stage: 32 rows x 64 bf16, 16B per thread
  const int krow = tid >> 3;
  const int kcol = (tid & 7) * 8;
  // V transposed stage: thread covers key = tid&31, d block = (tid>>5)*8
  const int vkey = tid & 31;
  const int vd0  = (tid >> 5) * 8;

  // ---- prologue: async Q tile, async K chunk 0, manual V^T chunk 0 ----
  {
    const int row = tid >> 1;
    const int c0  = (tid & 1) * 32;
#pragma unroll
    for (int t = 0; t < 4; ++t)
      async_copy16(Q + (size_t)row * DK_ + c0 + t * 8, &Qs[row][c0 + t * 8]);
  }
  async_copy16(Kp + (size_t)krow * DK_ + kcol, &Ks[0][krow][kcol]);
  {
    const uint4 raw = *(const uint4*)(Vp + (size_t)vkey * DK_ + vd0);
    const __bf16* e = (const __bf16*)&raw;
#pragma unroll
    for (int q = 0; q < 8; ++q) Vt[0][vd0 + q][vkey] = e[q];
  }
  wait_async0();
  __syncthreads();

  // Persistent Q A-fragments (DK=64 -> 2 chunks of K=32)
  FragBF16 qf[2];
#pragma unroll
  for (int c = 0; c < 2; ++c)
#pragma unroll
    for (int j = 0; j < 8; ++j) {
      const int kb = 32 * c + 2 * (j & 3) + ((j >> 2) << 4) + (half << 3);
      qf[c].u[j] = *(const unsigned int*)&Qs[wid * 16 + ln][kb];
    }

  float mrow[8], lrow[8];
  v8f cacc[4] = {};
#pragma unroll
  for (int r = 0; r < 8; ++r) { mrow[r] = -1e30f; lrow[r] = 0.0f; }

  const float scale = 0.125f;  // 1/sqrt(64)

  for (int kc = 0; kc < NC; ++kc) {
    const int cur = kc & 1;
    __syncthreads();
    // ---- stage next chunk into the other buffer while computing this one ----
    if (kc + 1 < NC) {
      const int nxt = cur ^ 1;
      async_copy16(Kp + (size_t)((kc + 1) * 32 + krow) * DK_ + kcol,
                   &Ks[nxt][krow][kcol]);
      const uint4 raw = *(const uint4*)(Vp + (size_t)((kc + 1) * 32 + vkey) * DK_ + vd0);
      const __bf16* e = (const __bf16*)&raw;
#pragma unroll
      for (int q = 0; q < 8; ++q) Vt[nxt][vd0 + q][vkey] = e[q];
    }

    // ---- scores: s0 = keys 0..15, s1 = keys 16..31 of this chunk ----
    v8f s0a = {}, s1a = {};
#pragma unroll
    for (int c = 0; c < 2; ++c) {
      FragBF16 kf0, kf1;
#pragma unroll
      for (int j = 0; j < 8; ++j) {
        const int kb = 32 * c + 2 * (j & 3) + ((j >> 2) << 4) + (half << 3);
        kf0.u[j] = *(const unsigned int*)&Ks[cur][ln][kb];
        kf1.u[j] = *(const unsigned int*)&Ks[cur][16 + ln][kb];
      }
      s0a = __builtin_amdgcn_wmma_f32_16x16x32_bf16(false, qf[c].v, false, kf0.v,
                                                    (short)0, s0a, false, false);
      s1a = __builtin_amdgcn_wmma_f32_16x16x32_bf16(false, qf[c].v, false, kf1.v,
                                                    (short)0, s1a, false, false);
    }

    // ---- scale + mask + online softmax ----
    float s0[8], s1[8], rm[8];
#pragma unroll
    for (int r = 0; r < 8; ++r) {
      const int grow = qb0 + wid * 16 + r + 8 * half;
      const int gc   = kc * 32 + ln;
      const float mk0 = mask[(size_t)grow * S_ + gc];
      const float mk1 = mask[(size_t)grow * S_ + gc + 16];
      s0[r] = s0a[r] * scale - 1e9f * mk0;
      s1[r] = s1a[r] * scale - 1e9f * mk1;
      rm[r] = fmaxf(s0[r], s1[r]);
    }
#pragma unroll
    for (int off = 1; off < 16; off <<= 1)
#pragma unroll
      for (int r = 0; r < 8; ++r)
        rm[r] = fmaxf(rm[r], __shfl_xor(rm[r], off, 32));

    float alpha[8], psum[8];
#pragma unroll
    for (int r = 0; r < 8; ++r) {
      const float mnew = fmaxf(mrow[r], rm[r]);
      alpha[r] = __expf(mrow[r] - mnew);
      mrow[r]  = mnew;
      s0[r] = __expf(s0[r] - mnew);
      s1[r] = __expf(s1[r] - mnew);
      psum[r] = s0[r] + s1[r];
    }
#pragma unroll
    for (int off = 1; off < 16; off <<= 1)
#pragma unroll
      for (int r = 0; r < 8; ++r)
        psum[r] += __shfl_xor(psum[r], off, 32);
#pragma unroll
    for (int r = 0; r < 8; ++r) lrow[r] = lrow[r] * alpha[r] + psum[r];
#pragma unroll
    for (int t = 0; t < 4; ++t)
#pragma unroll
      for (int r = 0; r < 8; ++r) cacc[t][r] *= alpha[r];

    // ---- transpose P (C layout -> A layout) through per-wave LDS region ----
#pragma unroll
    for (int r = 0; r < 8; ++r) {
      Pt[wid][r + 8 * half][ln]      = (__bf16)s0[r];
      Pt[wid][r + 8 * half][16 + ln] = (__bf16)s1[r];
    }
    wait_ds0();  // wave-local store->gather ordering

    FragBF16 pf;
#pragma unroll
    for (int j = 0; j < 8; ++j) {
      const int kb = 2 * (j & 3) + ((j >> 2) << 4) + (half << 3);
      pf.u[j] = *(const unsigned int*)&Pt[wid][ln][kb];
    }
    // ctx += P[16,32] @ V[32,64]  (4 n-tiles of 16)
#pragma unroll
    for (int t = 0; t < 4; ++t) {
      FragBF16 vf;
#pragma unroll
      for (int j = 0; j < 8; ++j) {
        const int kb = 2 * (j & 3) + ((j >> 2) << 4) + (half << 3);
        vf.u[j] = *(const unsigned int*)&Vt[cur][t * 16 + ln][kb];
      }
      cacc[t] = __builtin_amdgcn_wmma_f32_16x16x32_bf16(false, pf.v, false, vf.v,
                                                        (short)0, cacc[t], false, false);
    }

    // drain my async K copies before the next-iteration publishing barrier
    wait_async0();
  }

  // ---- write ctx f32 in [B, S, H*DV] layout (ready for output projection) ----
#pragma unroll
  for (int t = 0; t < 4; ++t)
#pragma unroll
    for (int r = 0; r < 8; ++r) {
      const int grow = qb0 + wid * 16 + r + 8 * half;
      const int d    = t * 16 + ln;
      const float val = cacc[t][r] / lrow[r];
      ctx[((size_t)b * S_ + grow) * (H_ * DV_) + (size_t)h * DV_ + d] = val;
    }
}

extern "C" void kernel_launch(void* const* d_in, const int* in_sizes, int n_in,
                              void* d_out, int out_size, void* d_ws, size_t ws_size,
                              hipStream_t stream) {
  (void)in_sizes; (void)n_in; (void)out_size; (void)ws_size;

  const float* queries = (const float*)d_in[0];
  const float* keys    = (const float*)d_in[1];
  const float* values  = (const float*)d_in[2];
  const float* mask    = (const float*)d_in[3];
  const float* Wq = (const float*)d_in[4];
  const float* bq = (const float*)d_in[5];
  const float* Wk = (const float*)d_in[6];
  const float* bk = (const float*)d_in[7];
  const float* Wv = (const float*)d_in[8];
  const float* bv = (const float*)d_in[9];
  const float* Wo = (const float*)d_in[10];
  const float* bo = (const float*)d_in[11];
  float* out = (float*)d_out;

  // Workspace: q/k/v bf16 (16 MB each) + ctx f32 (32 MB) = 80 MB
  const size_t elems = (size_t)M_ * N_;
  unsigned short* qb = (unsigned short*)d_ws;
  unsigned short* kb = qb + elems;
  unsigned short* vb = kb + elems;
  float* ctx = (float*)(vb + elems);

  const dim3 gg(N_ / BN, M_ / BM, 1);
  gemm_wmma_kernel<true><<<gg, 256, 0, stream>>>(queries, Wq, bq, nullptr, qb);
  gemm_wmma_kernel<true><<<gg, 256, 0, stream>>>(keys,    Wk, bk, nullptr, kb);
  gemm_wmma_kernel<true><<<gg, 256, 0, stream>>>(values,  Wv, bv, nullptr, vb);

  attn_wmma_kernel<<<dim3(S_ / 128, H_, B_), 256, 0, stream>>>(qb, kb, vb, mask, ctx);

  gemm_wmma_kernel<false><<<gg, 256, 0, stream>>>(ctx, Wo, bo, out, nullptr);
}
